// Linear2MPO_87479893885512
// MI455X (gfx1250) — matrix operand order repaired and checked by reading, not compiled, not run on hardware
//
#include <hip/hip_runtime.h>
#include <stdint.h>

typedef __attribute__((ext_vector_type(16))) __bf16   v16bf;
typedef __attribute__((ext_vector_type(8)))  float    v8f;
typedef __attribute__((ext_vector_type(4)))  float    f32x4;
typedef __attribute__((ext_vector_type(4)))  uint32_t u32x4;

// ---------- bf16 helpers ----------
#if __has_builtin(__builtin_amdgcn_cvt_pk_bf16_f32)
typedef __attribute__((ext_vector_type(2))) __bf16 v2bf;
__device__ __forceinline__ uint32_t f2bf_pack(float lo, float hi) {
  v2bf r = __builtin_amdgcn_cvt_pk_bf16_f32(lo, hi);
  return __builtin_bit_cast(uint32_t, r);
}
#else
__device__ __forceinline__ uint32_t f2bf_pack(float lo, float hi) {
  uint32_t a = __builtin_bit_cast(uint32_t, lo);
  uint32_t b = __builtin_bit_cast(uint32_t, hi);
  a += 0x7FFFu + ((a >> 16) & 1u);   // round-to-nearest-even
  b += 0x7FFFu + ((b >> 16) & 1u);
  return (a >> 16) | (b & 0xFFFF0000u);
}
#endif

__device__ __forceinline__ uint16_t f2bf(float f) {
  uint32_t u = __builtin_bit_cast(uint32_t, f);
  u += 0x7FFFu + ((u >> 16) & 1u);
  return (uint16_t)(u >> 16);
}

// ---------- TT contraction step, vectorized over 4 consecutive q ----------
// t[I,O,R] x core[R,i,o,Q] -> out[(I*i),(O*o),Q]; requires Q % 4 == 0.
// Matches reference einsum('IOr,rioq->IiOoq') + reshape(I*i, O*o, q).
__global__ void mpo_step_q4_kernel(const float* __restrict__ t,
                                   const float* __restrict__ core,
                                   float* __restrict__ out,
                                   int O, int R, int i_dim, int o_dim, int Q,
                                   int total4) {
  int idx = blockIdx.x * blockDim.x + threadIdx.x;
  if (idx >= total4) return;
  const int Q4 = Q >> 2;
  int q4   = idx % Q4;
  int rest = idx / Q4;
  int Oo   = O * o_dim;
  int On   = rest % Oo;
  int In   = rest / Oo;
  int Iold = In / i_dim, ik = In % i_dim;
  int Oold = On / o_dim, ok = On % o_dim;
  const float* tp = t + (size_t)(Iold * O + Oold) * R;
  const int cstride = i_dim * o_dim * Q;
  const float* cp = core + (size_t)(ik * o_dim + ok) * Q + (q4 << 2);
  f32x4 acc = {0.f, 0.f, 0.f, 0.f};
  #pragma unroll 4
  for (int r = 0; r < R; ++r) {
    const float tv = tp[r];
    const f32x4 cv = *(const f32x4*)(cp + (size_t)r * cstride);
    acc.x += tv * cv.x;
    acc.y += tv * cv.y;
    acc.z += tv * cv.z;
    acc.w += tv * cv.w;
  }
  *(f32x4*)(out + (size_t)rest * Q + (q4 << 2)) = acc;
}

// ---------- final step (core4) + pre-swizzle into WMMA B-fragment layout (bf16) ----------
// M[row=K(768)][col=N(3072)]; B fragment per ISA 7.12.2 (16-bit B, 32x16, wave32):
//   VGPR j, lanes 0-15 : K = 2j, 2j+1 (j<4), 2j+8, 2j+9 (j>=4)   (N = lane)
//   VGPR j, lanes 16-31: K = 8+2j,... (j<4), 16+2j,... (j>=4)  (N = lane-16)
// Stored as Bpk[kt][nt][lane][j] packed-bf16-pair dwords (kt: K/32, nt: N/16).
__global__ void mpo_final_pack_kernel(const float* __restrict__ t3,
                                      const float* __restrict__ core4,
                                      uint16_t* __restrict__ Bpk) {
  int idx = blockIdx.x * blockDim.x + threadIdx.x;   // over 768*3072
  if (idx >= 768 * 3072) return;
  int col = idx % 3072, row = idx / 3072;
  int Iold = row >> 2, ik = row & 3;
  int Oold = col >> 2, ok = col & 3;
  const float* tp = t3 + (size_t)(Iold * 768 + Oold) * 16;
  const float* cp = core4 + (ik * 4 + ok);           // stride per r = i*o*Q = 16
  float acc = 0.0f;
  #pragma unroll
  for (int r = 0; r < 16; ++r) acc += tp[r] * cp[r * 16];

  int kt   = row >> 5,  kk = row & 31;
  int half = (kk >> 3) & 1,   jj = ((kk & 7) + ((kk >> 4) << 3)) >> 1, p = kk & 1;
  int lane = half * 16 + (col & 15);
  int nt   = col >> 4;
  size_t off = ((((size_t)kt * 192 + nt) * 32 + lane) * 8 + jj) * 2 + p;
  Bpk[off] = f2bf(acc);
}

// ---------- GEMM building blocks ----------
__device__ __forceinline__ void load_b(u32x4 b[8], const uint32_t* bcol, int kt) {
  const uint32_t* bk = bcol + (size_t)kt * (192 * 32 * 8);
  #pragma unroll
  for (int t = 0; t < 4; ++t) {
    const u32x4* bp = (const u32x4*)(bk + (size_t)t * (32 * 8));
    b[2 * t]     = bp[0];
    b[2 * t + 1] = bp[1];
  }
}

__device__ __forceinline__ void load_a(f32x4 a[4], const float* xrow, int kt) {
  const float* xk = xrow + kt * 32;
  a[0] = *(const f32x4*)(xk);
  a[1] = *(const f32x4*)(xk + 4);
  a[2] = *(const f32x4*)(xk + 16);
  a[3] = *(const f32x4*)(xk + 20);
}

__device__ __forceinline__ void mma_step(v8f acc[4], const f32x4 a[4],
                                         const u32x4 b[8]) {
  union { uint32_t u[8]; v16bf v; } A;
  #pragma unroll
  for (int j = 0; j < 4; ++j) {
    A.u[2 * j]     = f2bf_pack(a[j].x, a[j].y);
    A.u[2 * j + 1] = f2bf_pack(a[j].z, a[j].w);
  }
  #pragma unroll
  for (int t = 0; t < 4; ++t) {
    union { u32x4 q[2]; v16bf v; } B;
    B.q[0] = b[2 * t];
    B.q[1] = b[2 * t + 1];
    acc[t] = __builtin_amdgcn_wmma_f32_16x16x32_bf16(
        /*neg_a=*/false, A.v, /*neg_b=*/false, B.v,
        /*c_mod=*/(short)0, acc[t], /*reuse_a=*/false, /*reuse_b=*/false);
  }
}

// ---------- GEMM: y[32768,3072] = x[32768,768] @ M + bias, bf16 WMMA / fp32 accum ----------
// Block = 256 threads = 8 waves. Wave tile: 16M x 64N (4 wmma accumulators).
// Block tile: 32M x 256N. Grid: (12, 1024) exactly tiles the problem ->
// EXEC all-1s at every WMMA (ISA requirement).
// A and B register buffers are double-buffered across K-steps (loop unrolled
// by 2, two explicit buffer sets), so the WMMAs for step k only depend on
// loads issued one iteration earlier and never wait for the in-flight loads
// of step k+1.
__global__ __launch_bounds__(256) void mpo_gemm_kernel(
    const float* __restrict__ x, const uint32_t* __restrict__ Bpk,
    const float* __restrict__ bias, float* __restrict__ y) {
  const int lane = threadIdx.x & 31;
  const int wave = threadIdx.x >> 5;
  const int lrow = lane & 15;
  const int half = lane >> 4;
  const int rowBase = blockIdx.y * 32 + (wave & 1) * 16;
  const int colBase = blockIdx.x * 256 + (wave >> 1) * 64;

  // A-matrix 16-bit layout (16x32): lanes 0-15 hold K 0-7 & 16-23,
  // lanes 16-31 hold K 8-15 & 24-31.
  const float* xrow = x + (size_t)(rowBase + lrow) * 768 + half * 8;
  const uint32_t* bcol = Bpk + ((size_t)(colBase >> 4) * 32 + lane) * 8;

  v8f acc[4];
  #pragma unroll
  for (int t = 0; t < 4; ++t)
    acc[t] = (v8f){0.f, 0.f, 0.f, 0.f, 0.f, 0.f, 0.f, 0.f};

  u32x4 bEven[8], bOdd[8];
  f32x4 aEven[4], aOdd[4];

  load_b(bEven, bcol, 0);
  load_a(aEven, xrow, 0);

  for (int kt = 0; kt < 24; kt += 2) {
    // prefetch odd step while even step computes
    load_b(bOdd, bcol, kt + 1);
    load_a(aOdd, xrow, kt + 1);
    mma_step(acc, aEven, bEven);

    // prefetch next even step while odd step computes (clamped, branch-free)
    const int kt2 = (kt + 2 < 24) ? (kt + 2) : 23;
    load_b(bEven, bcol, kt2);
    load_a(aEven, xrow, kt2);
    mma_step(acc, aOdd, bOdd);
  }

  // D layout: VGPR v -> M = v + 8*half, N = lane&15.
  // y is streamed write-once (403 MB) -> nontemporal stores keep x/B in L2.
  #pragma unroll
  for (int t = 0; t < 4; ++t) {
    const int col = colBase + t * 16 + lrow;
    const float bv = bias[col];
    float* yp = y + (size_t)(rowBase + half * 8) * 3072 + col;
    #pragma unroll
    for (int v = 0; v < 8; ++v)
      __builtin_nontemporal_store(acc[t][v] + bv, yp + (size_t)v * 3072);
  }
}

// ---------- launch ----------
extern "C" void kernel_launch(void* const* d_in, const int* in_sizes, int n_in,
                              void* d_out, int out_size, void* d_ws, size_t ws_size,
                              hipStream_t stream) {
  const float* x     = (const float*)d_in[0];
  const float* core0 = (const float*)d_in[1];   // [1,3,4,12]
  const float* core1 = (const float*)d_in[2];   // [12,4,4,192]
  const float* core2 = (const float*)d_in[3];   // [192,4,8,384]
  const float* core3 = (const float*)d_in[4];   // [384,4,6,16]
  const float* core4 = (const float*)d_in[5];   // [16,4,4,1]
  const float* bias  = (const float*)d_in[6];   // [3072]
  float* y = (float*)d_out;                     // [32768, 3072] fp32

  // workspace layout (floats): t1 | t2 | t3 | Bpk(bf16)  (~23.7 MB total)
  float*    t1  = (float*)d_ws;                 // 12*16*192   = 36864
  float*    t2  = t1 + 36864;                   // 48*128*384  = 2359296
  float*    t3  = t2 + 2359296;                 // 192*768*16  = 2359296
  uint16_t* Bpk = (uint16_t*)(t3 + 2359296);    // 768*3072 bf16 (pre-swizzled)

  // step 1: core0[0] as t0 [3,4,12]  x core1 -> t1 [12,16,192]
  {
    int total4 = (12 * 16 * 192) / 4;
    mpo_step_q4_kernel<<<(total4 + 255) / 256, 256, 0, stream>>>(
        core0, core1, t1, /*O=*/4, /*R=*/12, /*i=*/4, /*o=*/4, /*Q=*/192, total4);
  }
  // step 2: t1 [12,16,192] x core2 -> t2 [48,128,384]
  {
    int total4 = (48 * 128 * 384) / 4;
    mpo_step_q4_kernel<<<(total4 + 255) / 256, 256, 0, stream>>>(
        t1, core2, t2, /*O=*/16, /*R=*/192, /*i=*/4, /*o=*/8, /*Q=*/384, total4);
  }
  // step 3: t2 [48,128,384] x core3 -> t3 [192,768,16]
  {
    int total4 = (192 * 768 * 16) / 4;
    mpo_step_q4_kernel<<<(total4 + 255) / 256, 256, 0, stream>>>(
        t2, core3, t3, /*O=*/128, /*R=*/384, /*i=*/4, /*o=*/6, /*Q=*/16, total4);
  }
  // step 4: t3 x core4 -> M[768,3072] -> bf16 B-fragments
  mpo_final_pack_kernel<<<(768 * 3072 + 255) / 256, 256, 0, stream>>>(t3, core4, Bpk);

  // GEMM: 32768 x 3072 = x @ M + bias
  dim3 grid(3072 / 256, 32768 / 32);            // (12, 1024)
  mpo_gemm_kernel<<<grid, 256, 0, stream>>>(x, (const uint32_t*)Bpk, bias, y);
}
